// GCL_2362232013237
// MI455X (gfx1250) — matrix operand order repaired and checked by reading, not compiled, not run on hardware
//
#include <hip/hip_runtime.h>
#include <hip/hip_bf16.h>
#include <math.h>

// ---------- model constants ----------
#define NHEADS 8
#define HDIM   32
#define CDIM   256
#define HH     48
#define WW     64
#define NTOK   (HH*WW)        // 3072
#define KWIN   13
#define KK2    (KWIN*KWIN)    // 169
#define DEPTH  6

typedef __attribute__((ext_vector_type(16))) _Float16 v16h;
typedef __attribute__((ext_vector_type(8)))  _Float16 v8h;
typedef __attribute__((ext_vector_type(8)))  float    v8f;

// ---------- NATTEN window helpers (K=13, D=3, nb=6) ----------
__device__ __forceinline__ int win_start(int i, int L) {
    int imodd = i % 3;
    int a = (L / 3) * 3;
    int b = L - a;
    int right = (imodd < b) ? (L - b + imodd - 36) : (a + imodd - 39);
    if (i - 18 < 0) return imodd;
    if (i + 18 >= L) return right;
    return i - 18;
}
__device__ __forceinline__ int pb_start(int i, int L) {
    if (i < 18) return 12 - i / 3;
    if (i + 18 >= L) return (L - i - 1) / 3;
    return 6;
}

// ---------- pack weight matrix (Nout x Kd, row-major f32) into WMMA B-fragment layout ----------
// packed[((tn*ntK + tk)*32 + lane)*16 + i] = W[n][k],
//   n = tn*16 + (lane&15), k = tk*32 + 16*(lane>>4) + i  (contiguous 16 halves per lane)
__global__ __launch_bounds__(256) void pack_w_kernel(const float* __restrict__ W,
                                                     _Float16* __restrict__ out,
                                                     int Nout, int Kd) {
    int idx = blockIdx.x * 256 + threadIdx.x;
    if (idx >= Nout * Kd) return;
    int i    = idx & 15;
    int lane = (idx >> 4) & 31;
    int tile = idx >> 9;
    int ntK  = Kd >> 5;
    int tk   = tile % ntK;
    int tn   = tile / ntK;
    int n = (tn << 4) + (lane & 15);
    int k = (tk << 5) + ((lane >> 4) << 4) + i;
    out[idx] = (_Float16)W[n * Kd + k];
}

// ---------- (C,H,W) -> (tok, C) ----------
__global__ __launch_bounds__(256) void to_tokens_kernel(const float* __restrict__ x,
                                                        float* __restrict__ t) {
    int idx = blockIdx.x * 256 + threadIdx.x;   // over CDIM*NTOK
    int c   = idx / NTOK;
    int pix = idx - c * NTOK;
    t[pix * CDIM + c] = x[idx];
}

// ---------- LayerNorm over C=256; block = 256 threads = 1 token ----------
__global__ __launch_bounds__(256) void ln_kernel(const float* __restrict__ in,
                                                 const float* __restrict__ w,
                                                 const float* __restrict__ b,
                                                 _Float16* __restrict__ outh,
                                                 float* __restrict__ outf,
                                                 int transp) {
    __shared__ float red[256];
    int pix = blockIdx.x;
    int c   = threadIdx.x;
    float v = in[pix * CDIM + c];
    red[c] = v;
    __syncthreads();
    for (int s = 128; s > 0; s >>= 1) { if (c < s) red[c] += red[c + s]; __syncthreads(); }
    float mean = red[0] * (1.f / 256.f);
    __syncthreads();
    float dv = v - mean;
    red[c] = dv * dv;
    __syncthreads();
    for (int s = 128; s > 0; s >>= 1) { if (c < s) red[c] += red[c + s]; __syncthreads(); }
    float var = red[0] * (1.f / 256.f);
    float y = dv * rsqrtf(var + 1e-5f) * w[c] + b[c];
    if (outh) outh[pix * CDIM + c] = (_Float16)y;
    if (outf) outf[transp ? (c * NTOK + pix) : (pix * CDIM + c)] = y;
}

// ---------- WMMA GEMM: out = A(MxKd,f16) * Bpacked(KdxNd) + bias (+resid) (opt GELU) ----------
// one wave computes a 16(M) x 64(N) strip: 4 accumulators reuse one A fragment per K-step,
// cutting VMEM issue per WMMA from 4 loads to 2.5. Wave-uniform guard keeps EXEC all-ones.
__global__ __launch_bounds__(256) void gemm_wmma_kernel(
    const _Float16* __restrict__ A, const _Float16* __restrict__ Bp,
    const float* __restrict__ bias, const float* __restrict__ resid,
    float* __restrict__ outF, _Float16* __restrict__ outH,
    int M, int Kd, int Nd, int act) {
    int wave = threadIdx.x >> 5;
    int lane = threadIdx.x & 31;
    int ntN4 = Nd >> 6;                       // N strips of 64
    int ntM  = M >> 4;
    int strip = blockIdx.x * 8 + wave;
    if (strip >= ntM * ntN4) return;          // uniform per wave
    int mt  = strip / ntN4;
    int nt0 = (strip % ntN4) << 2;            // first of 4 consecutive 16-wide N tiles
    int mlo = lane & 15;
    int hlf = lane >> 4;
    int ntK = Kd >> 5;
    const _Float16* arow = A + (size_t)((mt << 4) + mlo) * Kd + hlf * 16;
    const v16h* bt0 = (const v16h*)Bp + (size_t)(nt0 + 0) * ntK * 32 + lane;
    const v16h* bt1 = (const v16h*)Bp + (size_t)(nt0 + 1) * ntK * 32 + lane;
    const v16h* bt2 = (const v16h*)Bp + (size_t)(nt0 + 2) * ntK * 32 + lane;
    const v16h* bt3 = (const v16h*)Bp + (size_t)(nt0 + 3) * ntK * 32 + lane;

    v8f c0 = {}, c1 = {}, c2 = {}, c3 = {};
    for (int tk = 0; tk < ntK; ++tk) {
        v8h a0 = *(const v8h*)(arow + tk * 32);
        v8h a1 = *(const v8h*)(arow + tk * 32 + 8);
        v16h a;
#pragma unroll
        for (int i = 0; i < 8; ++i) { a[i] = a0[i]; a[i + 8] = a1[i]; }
        v16h b0 = bt0[tk * 32];
        v16h b1 = bt1[tk * 32];
        v16h b2 = bt2[tk * 32];
        v16h b3 = bt3[tk * 32];
        c0 = __builtin_amdgcn_wmma_f32_16x16x32_f16(false, a, false, b0, (short)0, c0, false, false);
        c1 = __builtin_amdgcn_wmma_f32_16x16x32_f16(false, a, false, b1, (short)0, c1, false, false);
        c2 = __builtin_amdgcn_wmma_f32_16x16x32_f16(false, a, false, b2, (short)0, c2, false, false);
        c3 = __builtin_amdgcn_wmma_f32_16x16x32_f16(false, a, false, b3, (short)0, c3, false, false);
    }
#pragma unroll
    for (int t = 0; t < 4; ++t) {
        v8f c = (t == 0) ? c0 : (t == 1) ? c1 : (t == 2) ? c2 : c3;
        int n = ((nt0 + t) << 4) + mlo;
        float bn = bias ? bias[n] : 0.f;
#pragma unroll
        for (int r = 0; r < 8; ++r) {
            int mrow = (mt << 4) + r + (hlf << 3);
            float v = c[r] + bn;
            if (resid) v += resid[(size_t)mrow * Nd + n];
            if (act)   v = 0.5f * v * (1.f + erff(v * 0.70710678118654752f));
            if (outF) outF[(size_t)mrow * Nd + n] = v;
            if (outH) outH[(size_t)mrow * Nd + n] = (_Float16)v;
        }
    }
}

// ---------- dilated neighborhood attention: block = 1 pixel, wave = 1 head ----------
__global__ __launch_bounds__(256) void natten_kernel(
    const float* __restrict__ qkv,      // (NTOK, 768) : [q|k|v] each (head,32)
    const float* __restrict__ rpb,      // (NHEADS, 25, 25)
    const float* __restrict__ lrm,      // (13,13)
    _Float16* __restrict__ out) {       // (NTOK, 256)
    int pix  = blockIdx.x;
    int r    = pix / WW;
    int ccol = pix - r * WW;
    int head = threadIdx.x >> 5;
    int lane = threadIdx.x & 31;

    __shared__ float qs[NHEADS][HDIM];
    __shared__ float sc[NHEADS][176];

    const float scale = 0.17677669529663687f;   // 32^-0.5
    qs[head][lane] = qkv[(size_t)pix * 768 + head * HDIM + lane] * scale;
    __syncthreads();

    float qreg[HDIM];
#pragma unroll
    for (int i = 0; i < HDIM; ++i) qreg[i] = qs[head][i];

    int sh = win_start(r, HH), sw = win_start(ccol, WW);
    int ph = pb_start(r, HH),  pw = pb_start(ccol, WW);
    const float* rp = rpb + head * 625;

    // phase 1: scores (lane-per-neighbor), (q.k + bias) * (gauss + lrm/SC)
    float lmax = -1e30f;
    for (int nb = lane; nb < KK2; nb += 32) {
        int i = nb / KWIN, j = nb - i * KWIN;
        int kr = sh + 3 * i, kc = sw + 3 * j;
        const float* kv = qkv + (size_t)(kr * WW + kc) * 768 + CDIM + head * HDIM;
        float dot = 0.f;
#pragma unroll
        for (int d = 0; d < HDIM; ++d) dot = fmaf(qreg[d], kv[d], dot);
        float bia = rp[(ph + i) * 25 + (pw + j)];
        float di = (float)(i - 6), dj = (float)(j - 6);
        float ker = expf(-(di * di + dj * dj) * (1.f / 162.f)) + lrm[nb] * 10.f;
        float s = (dot + bia) * ker;
        sc[head][nb] = s;
        lmax = fmaxf(lmax, s);
    }
    for (int off = 16; off > 0; off >>= 1) lmax = fmaxf(lmax, __shfl_xor(lmax, off, 32));
    float lsum = 0.f;
    for (int nb = lane; nb < KK2; nb += 32) {
        float e = expf(sc[head][nb] - lmax);
        sc[head][nb] = e;
        lsum += e;
    }
    for (int off = 16; off > 0; off >>= 1) lsum += __shfl_xor(lsum, off, 32);
    float inv = 1.f / lsum;

    // phase 2: output (lane-per-dim), coalesced v loads
    float acc = 0.f;
    for (int nb = 0; nb < KK2; ++nb) {
        int i = nb / KWIN, j = nb - i * KWIN;
        int kr = sh + 3 * i, kc = sw + 3 * j;
        float wgt = sc[head][nb] * inv;
        acc = fmaf(wgt, qkv[(size_t)(kr * WW + kc) * 768 + 2 * CDIM + head * HDIM + lane], acc);
    }
    out[(size_t)pix * CDIM + head * HDIM + lane] = (_Float16)acc;
}

// ---------- launch ----------
extern "C" void kernel_launch(void* const* d_in, const int* in_sizes, int n_in,
                              void* d_out, int out_size, void* d_ws, size_t ws_size,
                              hipStream_t stream) {
    const float* x      = (const float*)d_in[0];
    const float* ln1_w  = (const float*)d_in[1];
    const float* ln1_b  = (const float*)d_in[2];
    const float* qkv_w  = (const float*)d_in[3];
    const float* qkv_b  = (const float*)d_in[4];
    const float* rpb    = (const float*)d_in[5];
    const float* lr_m   = (const float*)d_in[6];
    const float* proj_w = (const float*)d_in[7];
    const float* proj_b = (const float*)d_in[8];
    const float* ln2_w  = (const float*)d_in[9];
    const float* ln2_b  = (const float*)d_in[10];
    const float* fc1_w  = (const float*)d_in[11];
    const float* fc1_b  = (const float*)d_in[12];
    const float* fc2_w  = (const float*)d_in[13];
    const float* fc2_b  = (const float*)d_in[14];
    const float* nf_w   = (const float*)d_in[15];
    const float* nf_b   = (const float*)d_in[16];
    float* outp = (float*)d_out;

    char* p = (char*)d_ws;
    float*     tbuf    = (float*)p;     p += (size_t)NTOK * CDIM * 4;        // 3 MB
    _Float16*  lnout   = (_Float16*)p;  p += (size_t)NTOK * CDIM * 2;        // 1.5 MB
    float*     qkvbuf  = (float*)p;     p += (size_t)NTOK * 3 * CDIM * 4;    // 9 MB
    _Float16*  attnout = (_Float16*)p;  p += (size_t)NTOK * CDIM * 2;        // 1.5 MB
    _Float16*  hid     = (_Float16*)p;  p += (size_t)NTOK * 4 * CDIM * 2;    // 6 MB
    _Float16*  qkvP    = (_Float16*)p;  p += (size_t)DEPTH * 3 * CDIM * CDIM * 2;
    _Float16*  projP   = (_Float16*)p;  p += (size_t)DEPTH * CDIM * CDIM * 2;
    _Float16*  fc1P    = (_Float16*)p;  p += (size_t)DEPTH * 4 * CDIM * CDIM * 2;
    _Float16*  fc2P    = (_Float16*)p;  p += (size_t)DEPTH * CDIM * 4 * CDIM * 2;

    // pack all weights into WMMA fragment layout
    for (int l = 0; l < DEPTH; ++l) {
        pack_w_kernel<<<(3 * CDIM * CDIM + 255) / 256, 256, 0, stream>>>(
            qkv_w + (size_t)l * 3 * CDIM * CDIM, qkvP + (size_t)l * 3 * CDIM * CDIM, 3 * CDIM, CDIM);
        pack_w_kernel<<<(CDIM * CDIM + 255) / 256, 256, 0, stream>>>(
            proj_w + (size_t)l * CDIM * CDIM, projP + (size_t)l * CDIM * CDIM, CDIM, CDIM);
        pack_w_kernel<<<(4 * CDIM * CDIM + 255) / 256, 256, 0, stream>>>(
            fc1_w + (size_t)l * 4 * CDIM * CDIM, fc1P + (size_t)l * 4 * CDIM * CDIM, 4 * CDIM, CDIM);
        pack_w_kernel<<<(4 * CDIM * CDIM + 255) / 256, 256, 0, stream>>>(
            fc2_w + (size_t)l * 4 * CDIM * CDIM, fc2P + (size_t)l * 4 * CDIM * CDIM, CDIM, 4 * CDIM);
    }

    to_tokens_kernel<<<(CDIM * NTOK) / 256, 256, 0, stream>>>(x, tbuf);

    const int ntM = NTOK / 16;                 // 192
    for (int l = 0; l < DEPTH; ++l) {
        // LN1 -> f16
        ln_kernel<<<NTOK, 256, 0, stream>>>(tbuf, ln1_w + l * CDIM, ln1_b + l * CDIM,
                                            lnout, nullptr, 0);
        // QKV: 3072x256 * 256x768
        {
            int strips = ntM * (3 * CDIM / 64);
            gemm_wmma_kernel<<<(strips + 7) / 8, 256, 0, stream>>>(
                lnout, qkvP + (size_t)l * 3 * CDIM * CDIM, qkv_b + l * 3 * CDIM,
                nullptr, qkvbuf, nullptr, NTOK, CDIM, 3 * CDIM, 0);
        }
        // neighborhood attention
        natten_kernel<<<NTOK, 256, 0, stream>>>(qkvbuf, rpb + (size_t)l * NHEADS * 625,
                                                lr_m + (size_t)l * KK2, attnout);
        // proj + bias + residual -> tbuf
        {
            int strips = ntM * (CDIM / 64);
            gemm_wmma_kernel<<<(strips + 7) / 8, 256, 0, stream>>>(
                attnout, projP + (size_t)l * CDIM * CDIM, proj_b + l * CDIM,
                tbuf, tbuf, nullptr, NTOK, CDIM, CDIM, 0);
        }
        // LN2 -> f16
        ln_kernel<<<NTOK, 256, 0, stream>>>(tbuf, ln2_w + l * CDIM, ln2_b + l * CDIM,
                                            lnout, nullptr, 0);
        // FC1 + GELU -> hid (f16)
        {
            int strips = ntM * (4 * CDIM / 64);
            gemm_wmma_kernel<<<(strips + 7) / 8, 256, 0, stream>>>(
                lnout, fc1P + (size_t)l * 4 * CDIM * CDIM, fc1_b + l * 4 * CDIM,
                nullptr, nullptr, hid, NTOK, CDIM, 4 * CDIM, 1);
        }
        // FC2 + bias + residual -> tbuf
        {
            int strips = ntM * (CDIM / 64);
            gemm_wmma_kernel<<<(strips + 7) / 8, 256, 0, stream>>>(
                hid, fc2P + (size_t)l * 4 * CDIM * CDIM, fc2_b + l * CDIM,
                tbuf, tbuf, nullptr, NTOK, 4 * CDIM, CDIM, 0);
        }
    }
    // final LN fused with tokens -> NCHW transpose
    ln_kernel<<<NTOK, 256, 0, stream>>>(tbuf, nf_w, nf_b, nullptr, outp, 1);
}
